// BlockCrossAttention_10806137717136
// MI455X (gfx1250) — compile-verified, matrix operand order checked by reading, not run on hardware
//
#include <hip/hip_runtime.h>

// ---------------- problem constants (from reference) ----------------
#define CB   2
#define CL   8192
#define CS   4096
#define CD   1024
#define CH   16
#define CHKV 4
#define CHD  64
#define CBS  16
#define CNB  512              // L / BS, no padding since 8192 % 16 == 0
#define CSCALE 0.125f         // 1/sqrt(64)
#define NEG_INF (-__builtin_inff())

// gfx1250 async global->LDS (ASYNCcnt) if this toolchain exposes the builtins
#if defined(__has_builtin)
#  if __has_builtin(__builtin_amdgcn_global_load_async_to_lds_b128) && \
      __has_builtin(__builtin_amdgcn_s_wait_asynccnt)
#    define USE_ASYNC_LDS 1
#  endif
#endif
#ifndef USE_ASYNC_LDS
#  define USE_ASYNC_LDS 0
#endif

typedef __attribute__((ext_vector_type(16))) _Float16 v16h;
typedef __attribute__((ext_vector_type(8)))  _Float16 v8h;
typedef __attribute__((ext_vector_type(8)))  float    v8f;
typedef int i4 __attribute__((vector_size(16)));   // builtin's b128 operand type

// LDS row pitch for staged B chunks: 32 data halves padded to 40 (80B) so the
// 16-lane phases of ds_read_b128 hit distinct bank quads (gcd(20,64)=4).
#define BPITCH 40

// ---------------------------------------------------------------------------
// WMMA fragment load. Convention (CDNA5 ISA, 16-bit A 16x32):
//   lane<16 : row = row0+lane,    K in {k0+0..7,  k0+16..23}
//   lane>=16: row = row0+lane-16, K in {k0+8..15, k0+24..31}
// B operands use the same pattern on the transposed matrix ([N,K] storage).
// Each lane = two contiguous 16B chunks -> global_load_b128 x2 (or ds_read).
// ---------------------------------------------------------------------------
__device__ __forceinline__ v16h load_frag(const _Float16* __restrict__ base,
                                          int ld, int row0, int k0) {
  const int lane = threadIdx.x & 31;
  const int half = lane >> 4;
  const int r = row0 + (lane & 15);
  const _Float16* p = base + (size_t)r * ld + k0 + half * 8;
  v8h lo = *(const v8h*)(p);
  v8h hi = *(const v8h*)(p + 16);
  v16h f;
#pragma unroll
  for (int i = 0; i < 8; ++i) { f[i] = lo[i]; f[i + 8] = hi[i]; }
  return f;
}

// Same fragment pattern out of a staged LDS chunk (64 rows x 32 halves, BPITCH)
__device__ __forceinline__ v16h lds_frag(const _Float16* s, int row0) {
  const int lane = threadIdx.x & 31;
  const int half = lane >> 4;
  const _Float16* p = s + (row0 + (lane & 15)) * BPITCH + half * 8;
  v8h lo = *(const v8h*)(p);
  v8h hi = *(const v8h*)(p + 16);
  v16h f;
#pragma unroll
  for (int i = 0; i < 8; ++i) { f[i] = lo[i]; f[i + 8] = hi[i]; }
  return f;
}

__device__ __forceinline__ v8f wmma_f16(v16h a, v16h b, v8f c) {
  // v_wmma_f32_16x16x32_f16 : D = A(16x32 f16) x B(32x16 f16) + C(16x16 f32)
  return __builtin_amdgcn_wmma_f32_16x16x32_f16(false, a, false, b,
                                                (short)0, c, false, false);
}

// ---------------- elementwise helper kernels ----------------
__global__ void cast_f16_kernel(const float* __restrict__ src,
                                _Float16* __restrict__ dst, size_t n) {
  size_t i = (size_t)blockIdx.x * blockDim.x + threadIdx.x;
  if (i < n) dst[i] = (_Float16)src[i];
}

// src[R,C] f32  ->  dst[C,R] f16   (weights stored transposed: [N,K])
__global__ void cast_transpose_kernel(const float* __restrict__ src,
                                      _Float16* __restrict__ dst,
                                      int R, int C) {
  size_t i = (size_t)blockIdx.x * blockDim.x + threadIdx.x;
  if (i >= (size_t)R * C) return;
  int r = (int)(i % R);
  int c = (int)(i / R);
  dst[i] = (_Float16)src[(size_t)r * C + c];
}

// pooled16[b*NB+qb, d] = mean over 16 decoder tokens (f16 out)
__global__ void pool_kernel(const float* __restrict__ x,
                            _Float16* __restrict__ pooled) {
  size_t i = (size_t)blockIdx.x * blockDim.x + threadIdx.x;
  if (i >= (size_t)CB * CNB * CD) return;
  int d   = (int)(i % CD);
  int row = (int)(i / CD);
  int b = row / CNB, qb = row % CNB;
  const float* p = x + ((size_t)b * CL + (size_t)qb * CBS) * CD + d;
  float s = 0.f;
#pragma unroll
  for (int t = 0; t < CBS; ++t) s += p[(size_t)t * CD];
  pooled[i] = (_Float16)(s * (1.f / CBS));
}

// v16[b*S+s, g*HD+hd] -> vt16[((b*HKV+g)*HD + hd)*S + s]
__global__ void vtrans_kernel(const _Float16* __restrict__ vsrc,
                              _Float16* __restrict__ vt) {
  size_t i = (size_t)blockIdx.x * blockDim.x + threadIdx.x;
  if (i >= (size_t)CB * CHKV * CHD * CS) return;
  int s  = (int)(i % CS);  size_t rem = i / CS;
  int hd = (int)(rem % CHD); rem /= CHD;
  int g  = (int)(rem % CHKV);
  int b  = (int)(rem / CHKV);
  vt[i] = vsrc[((size_t)b * CS + s) * (CHKV * CHD) + g * CHD + hd];
}

// out[b,l,d] = y[b*NB + l/16, d]   (broadcast block result to its tokens)
__global__ void broadcast_kernel(const float* __restrict__ y,
                                 float* __restrict__ out) {
  size_t i = (size_t)blockIdx.x * blockDim.x + threadIdx.x;
  if (i >= (size_t)CB * CL * CD) return;
  int d = (int)(i % CD); size_t rem = i / CD;
  int l = (int)(rem % CL);
  int b = (int)(rem / CL);
  out[i] = y[((size_t)b * CNB + (l >> 4)) * CD + d];
}

// ---------------------------------------------------------------------------
// Block-cooperative WMMA GEMM: C[M,N] = A[M,K] (f16, lda) x Bt[N,K]^T (f16).
// 256 threads = 8 waves compute a 128x64 tile. The shared 64x32 B chunk is
// double-buffered in LDS (staged with gfx1250 async global->LDS, ASYNCcnt)
// and read by all 8 waves; A fragments are loaded directly per wave.
// Requires M % 128 == 0, N % 64 == 0, K % 32 == 0.
// OutT (branch-free epilogue): _Float16 or float.
// ---------------------------------------------------------------------------
template <typename OutT>
__global__ void __launch_bounds__(256)
gemm_wmma_kernel(const _Float16* __restrict__ A,
                 const _Float16* __restrict__ Bt,
                 OutT* __restrict__ Cout,
                 int M, int N, int K,
                 int lda, int ldb, int ldc) {
  __shared__ _Float16 sB[2][64 * BPITCH];     // 2 x 5 KB

  const int nstrips = N >> 6;
  const int bm = blockIdx.x / nstrips;        // 128-row block
  const int ns = blockIdx.x % nstrips;        // 64-col strip
  const int wv = threadIdx.x >> 5;
  const int lane = threadIdx.x & 31;
  const int mt = bm * 128 + wv * 16;          // this wave's 16 rows

  // cooperative B staging: 64 rows x 64B; one 16B chunk per thread
  const int crow = threadIdx.x >> 2;          // 0..63
  const int cchk = threadIdx.x & 3;           // 0..3
  const _Float16* bsrc = Bt + (size_t)(ns * 64 + crow) * ldb + cchk * 8;
  _Float16* bdst0 = &sB[0][crow * BPITCH + cchk * 8];
  _Float16* bdst1 = &sB[1][crow * BPITCH + cchk * 8];

  auto stageB = [&](int buf, int k) {
    const _Float16* src = bsrc + (size_t)k * 32;
    _Float16* dst = buf ? bdst1 : bdst0;
#if USE_ASYNC_LDS
    __builtin_amdgcn_global_load_async_to_lds_b128((i4*)src, (i4*)dst, 0, 0);
#else
    *(v8h*)dst = *(const v8h*)src;
#endif
  };

  const int nk = K >> 5;
  stageB(0, 0);
#if USE_ASYNC_LDS
  __builtin_amdgcn_s_wait_asynccnt(0);
#endif
  __syncthreads();

  v8f acc[4] = {};
  for (int k = 0; k < nk; ++k) {
    v16h a = load_frag(A, lda, mt, k * 32);
    v16h bf[4];
#pragma unroll
    for (int t = 0; t < 4; ++t) bf[t] = lds_frag(&sB[k & 1][0], t * 16);
    if (k + 1 < nk) stageB((k + 1) & 1, k + 1);
    if (k + 2 < nk)  // warm L2/WGP$ for the A stream (global_prefetch_b8)
      __builtin_prefetch(A + (size_t)(mt + (lane & 15)) * lda + (k + 2) * 32, 0, 3);
#pragma unroll
    for (int t = 0; t < 4; ++t) acc[t] = wmma_f16(a, bf[t], acc[t]);
#if USE_ASYNC_LDS
    if (k + 1 < nk) __builtin_amdgcn_s_wait_asynccnt(0);
#endif
    __syncthreads();
  }

  // D layout: lane<16 -> col=lane, rows j ; lane>=16 -> col=lane-16, rows j+8
  const int col = lane & 15, half = lane >> 4;
  OutT* cbase = Cout + (size_t)(mt + 8 * half) * ldc + ns * 64 + col;
#pragma unroll
  for (int t = 0; t < 4; ++t)
#pragma unroll
    for (int j = 0; j < 8; ++j)
      cbase[(size_t)j * ldc + t * 16] = (OutT)acc[t][j];
}

// ---------------------------------------------------------------------------
// Fused GQA block cross-attention, transposed-flash formulation.
// Per wave: one (b, head) and one 16-query tile. We compute scores^T tiles
// (A = K rows, B = Q^T) so each lane owns ONE query column; online-softmax
// stats need only a shfl_xor(16) between the two half-wave lanes sharing a
// query. Two consecutive 16-key score^T tiles per lane are *exactly* the
// B-operand layout for the PV WMMA (O^T = V^T x P^T): probabilities feed the
// next WMMA straight from registers.
// ---------------------------------------------------------------------------
__global__ void attn_wmma_kernel(const _Float16* __restrict__ q16,  // [B*NB, H*HD]
                                 const _Float16* __restrict__ k16,  // [B*S, HKV*HD]
                                 const _Float16* __restrict__ vt16, // [B*HKV, HD, S]
                                 const int* __restrict__ mask,      // [B, S]
                                 _Float16* __restrict__ o16) {      // [B*NB, H*HD]
  const int bh = blockIdx.y;                 // 0..B*H-1
  const int b = bh >> 4, h = bh & 15;
  const int g = h >> 2;                      // hpg = H/HKV = 4
  const int qt = blockIdx.x * (blockDim.x >> 5) + (threadIdx.x >> 5);
  if (qt >= CNB / 16) return;
  const int lane = threadIdx.x & 31;
  const int half = lane >> 4;
  const int qr = qt * 16;

  const _Float16* qbase = q16 + (size_t)(b * CNB) * (CH * CHD) + h * CHD;
  const _Float16* kbase = k16 + (size_t)(b * CS) * (CHKV * CHD) + g * CHD;
  const _Float16* vbase = vt16 + (size_t)(b * CHKV + g) * CHD * CS;
  const int* mrow = mask + (size_t)b * CS;

  // Q^T as B operand of scores^T (lane column = query), K = HD = 64 -> 2 frags
  v16h bq0 = load_frag(qbase, CH * CHD, qr, 0);
  v16h bq1 = load_frag(qbase, CH * CHD, qr, 32);

  v8f  oacc[4] = {};                // O^T: 4 hd-tiles x 16 queries
  float m_run = NEG_INF, l_run = 0.f;

  for (int c = 0; c < CS / 32; ++c) {
    if (c + 1 < CS / 32)  // prefetch next chunk's K rows
      __builtin_prefetch(kbase + (size_t)((c + 1) * 32 + (lane & 15)) * (CHKV * CHD), 0, 3);
    v8f st[2];
#pragma unroll
    for (int t = 0; t < 2; ++t) {
      const int kr = c * 32 + t * 16;
      v16h ak0 = load_frag(kbase, CHKV * CHD, kr, 0);
      v16h ak1 = load_frag(kbase, CHKV * CHD, kr, 32);
      v8f s = {};
      s = wmma_f16(ak0, bq0, s);
      s = wmma_f16(ak1, bq1, s);
      st[t] = s;
    }
    // scale + key-validity mask ; per-lane local max (all values = my query)
    float mloc = NEG_INF;
#pragma unroll
    for (int t = 0; t < 2; ++t)
#pragma unroll
      for (int j = 0; j < 8; ++j) {
        const int key = c * 32 + t * 16 + j + 8 * half;
        float sv = (mrow[key] != 0) ? st[t][j] * CSCALE : NEG_INF;
        st[t][j] = sv;
        mloc = fmaxf(mloc, sv);
      }
    const float mpair = fmaxf(mloc, __shfl_xor(mloc, 16));
    const float m_new = fmaxf(m_run, mpair);
    if (m_new == NEG_INF) continue;          // whole prefix masked so far

    const float factor = (m_run == NEG_INF) ? 0.f : __expf(m_run - m_new);
    l_run *= factor;
#pragma unroll
    for (int t4 = 0; t4 < 4; ++t4)
#pragma unroll
      for (int j = 0; j < 8; ++j) oacc[t4][j] *= factor;

    // probabilities -> B operand for PV (layout matches by construction)
    v16h pB;
#pragma unroll
    for (int t = 0; t < 2; ++t)
#pragma unroll
      for (int j = 0; j < 8; ++j) {
        const float sv = st[t][j];
        const float p = (sv == NEG_INF) ? 0.f : __expf(sv - m_new);
        l_run += p;
        pB[t * 8 + j] = (_Float16)p;
      }
    // O^T += V^T(16hd x 32keys) x P^T(32keys x 16q), 4 hd tiles
#pragma unroll
    for (int t4 = 0; t4 < 4; ++t4) {
      v16h av = load_frag(vbase, CS, t4 * 16, c * 32);
      oacc[t4] = wmma_f16(av, pB, oacc[t4]);
    }
    m_run = m_new;
  }

  const float l_tot = l_run + __shfl_xor(l_run, 16);
  const float inv = (l_tot > 0.f) ? 1.f / l_tot : 0.f;
  // O^T D layout -> 8 contiguous hd per lane: one 16B f16 store per hd-tile
  const int q = lane & 15;
  _Float16* orow = o16 + (size_t)(b * CNB + qr + q) * (CH * CHD) + h * CHD;
#pragma unroll
  for (int t4 = 0; t4 < 4; ++t4) {
    v8h outv;
#pragma unroll
    for (int j = 0; j < 8; ++j) outv[j] = (_Float16)(oacc[t4][j] * inv);
    *(v8h*)(orow + t4 * 16 + 8 * half) = outv;
  }
}

// ---------------------------------------------------------------------------
extern "C" void kernel_launch(void* const* d_in, const int* in_sizes, int n_in,
                              void* d_out, int out_size, void* d_ws, size_t ws_size,
                              hipStream_t stream) {
  const float* x    = (const float*)d_in[0];   // [B,L,D]
  const float* enc  = (const float*)d_in[1];   // [B,S,D]
  const int*   mask = (const int*)  d_in[2];   // [B,S]
  const float* Wq   = (const float*)d_in[3];   // [D, H*HD]
  const float* Wk   = (const float*)d_in[4];   // [D, HKV*HD]
  const float* Wv   = (const float*)d_in[5];   // [D, HKV*HD]
  const float* Wo   = (const float*)d_in[6];   // [H*HD, D]
  float* out = (float*)d_out;                  // [B,L,D]

  char* ws = (char*)d_ws;
  size_t off = 0;
  auto alloc = [&](size_t bytes) -> void* {
    void* p = ws + off;
    off = (off + bytes + 255) & ~(size_t)255;
    return p;
  };
  _Float16* enc16    = (_Float16*)alloc((size_t)CB * CS * CD * 2);        // 16 MB
  _Float16* pooled16 = (_Float16*)alloc((size_t)CB * CNB * CD * 2);       // 2 MB
  _Float16* Wq16t    = (_Float16*)alloc((size_t)CD * CH * CHD * 2);       // 2 MB
  _Float16* Wk16t    = (_Float16*)alloc((size_t)CD * CHKV * CHD * 2);     // 0.5 MB
  _Float16* Wv16t    = (_Float16*)alloc((size_t)CD * CHKV * CHD * 2);     // 0.5 MB
  _Float16* Wo16t    = (_Float16*)alloc((size_t)CD * CH * CHD * 2);       // 2 MB
  _Float16* q16      = (_Float16*)alloc((size_t)CB * CNB * CH * CHD * 2); // 2 MB
  _Float16* k16      = (_Float16*)alloc((size_t)CB * CS * CHKV * CHD * 2);// 4 MB
  _Float16* v16      = (_Float16*)alloc((size_t)CB * CS * CHKV * CHD * 2);// 4 MB
  _Float16* vt16     = (_Float16*)alloc((size_t)CB * CHKV * CHD * CS * 2);// 4 MB
  _Float16* o16      = (_Float16*)alloc((size_t)CB * CNB * CH * CHD * 2); // 2 MB
  float*    y        = (float*)   alloc((size_t)CB * CNB * CD * 4);       // 4 MB
  (void)ws_size; (void)in_sizes; (void)n_in; (void)out_size;

  const int T = 256;
  auto blocks = [&](size_t n) { return (unsigned)((n + T - 1) / T); };

  // f16 staging of enc + transposed f16 weights
  cast_f16_kernel<<<blocks((size_t)CB * CS * CD), T, 0, stream>>>(enc, enc16, (size_t)CB * CS * CD);
  cast_transpose_kernel<<<blocks((size_t)CD * CH * CHD), T, 0, stream>>>(Wq, Wq16t, CD, CH * CHD);
  cast_transpose_kernel<<<blocks((size_t)CD * CHKV * CHD), T, 0, stream>>>(Wk, Wk16t, CD, CHKV * CHD);
  cast_transpose_kernel<<<blocks((size_t)CD * CHKV * CHD), T, 0, stream>>>(Wv, Wv16t, CD, CHKV * CHD);
  cast_transpose_kernel<<<blocks((size_t)CH * CHD * CD), T, 0, stream>>>(Wo, Wo16t, CH * CHD, CD);

  // block-average pooling of decoder tokens
  pool_kernel<<<blocks((size_t)CB * CNB * CD), T, 0, stream>>>(x, pooled16);

  // projections (block-cooperative WMMA GEMMs), f16 outputs
  {
    int M = CB * CNB, N = CH * CHD, K = CD;          // 1024 x 1024 x 1024
    gemm_wmma_kernel<_Float16><<<(M / 128) * (N / 64), 256, 0, stream>>>(
        pooled16, Wq16t, q16, M, N, K, CD, CD, N);
  }
  {
    int M = CB * CS, N = CHKV * CHD, K = CD;         // 8192 x 256 x 1024
    gemm_wmma_kernel<_Float16><<<(M / 128) * (N / 64), 256, 0, stream>>>(
        enc16, Wk16t, k16, M, N, K, CD, CD, N);
    gemm_wmma_kernel<_Float16><<<(M / 128) * (N / 64), 256, 0, stream>>>(
        enc16, Wv16t, v16, M, N, K, CD, CD, N);
  }
  vtrans_kernel<<<blocks((size_t)CB * CHKV * CHD * CS), T, 0, stream>>>(v16, vt16);

  // fused masked GQA attention: grid (q-tiles/4, B*H), 4 waves per block
  attn_wmma_kernel<<<dim3((CNB / 16) / 4, CB * CH), 128, 0, stream>>>(q16, k16, vt16, mask, o16);

  // output projection (f32 out), then broadcast block rows to tokens
  {
    int M = CB * CNB, N = CD, K = CH * CHD;          // 1024 x 1024 x 1024
    gemm_wmma_kernel<float><<<(M / 128) * (N / 64), 256, 0, stream>>>(
        o16, Wo16t, y, M, N, K, K, K, N);
  }
  broadcast_kernel<<<blocks((size_t)CB * CL * CD), T, 0, stream>>>(y, out);
}
